// FlashAttentionV2_29076928594213
// MI455X (gfx1250) — matrix operand order repaired and checked by reading, not compiled, not run on hardware
//
#include <hip/hip_runtime.h>
#include <hip/hip_bf16.h>

// FlashAttention-2 (no 1/sqrt(d) scale) for n=8192, d=32 fp32 on gfx1250.
// Transposed formulation: S^T = K*Q^T, O^T = V^T * P^T on v_wmma_f32_16x16x32_f16.
// Fast path: prep kernels write f16 K (row-major) and f16 V^T (32 x n) into d_ws,
// so the hot loop does only b128 f16 loads (no converts, no strided gathers).

typedef _Float16 v16h __attribute__((ext_vector_type(16)));
typedef _Float16 h8   __attribute__((ext_vector_type(8)));
typedef _Float16 h4   __attribute__((ext_vector_type(4)));
typedef _Float16 h2   __attribute__((ext_vector_type(2)));
typedef float    v8f  __attribute__((ext_vector_type(8)));

#define ATTN_D 32
#define L2E 1.44269504088896340736f

// Raw v_exp_f32: we don't need subnormal-accurate exp2 for softmax weights.
__device__ __forceinline__ float fast_exp2(float x) {
#if __has_builtin(__builtin_amdgcn_exp2f)
  return __builtin_amdgcn_exp2f(x);
#else
  return exp2f(x);
#endif
}

// ---------------- prep kernels ----------------

// f32 -> f16 row-major copy (K). 4 elements per thread.
__global__ void cvt_f16_kernel(const float* __restrict__ src, _Float16* __restrict__ dst) {
  const int i = (blockIdx.x * blockDim.x + threadIdx.x) * 4;
  float4 f = *reinterpret_cast<const float4*>(src + i);
  h4 o;
  o.x = (_Float16)f.x; o.y = (_Float16)f.y; o.z = (_Float16)f.z; o.w = (_Float16)f.w;
  *reinterpret_cast<h4*>(dst + i) = o;
}

// V (n x 32 f32) -> VT (32 x n f16). grid = (n/256, 32); coalesced writes.
__global__ void transpose_v_kernel(const float* __restrict__ V, _Float16* __restrict__ VT, int n) {
  const int r = blockIdx.x * blockDim.x + threadIdx.x;  // key row
  const int d = blockIdx.y;                             // feature
  VT[(size_t)d * n + r] = (_Float16)V[(size_t)r * ATTN_D + d];
}

// ---------------- main kernel (fast path, f16 K/VT from ws) ----------------

// A-layout fragment from a 32-half f16 row: elements 0..7 <- cols 8h..8h+7,
// elements 8..15 <- cols 8h+16..8h+23. Two global_load_b128.
__device__ __forceinline__ v16h load_rowA16(const _Float16* __restrict__ row, int h) {
  h8 a = reinterpret_cast<const h8*>(row + 8 * h)[0];
  h8 b = reinterpret_cast<const h8*>(row + 8 * h + 16)[0];
  return __builtin_shufflevector(a, b, 0, 1, 2, 3, 4, 5, 6, 7, 8, 9, 10, 11, 12, 13, 14, 15);
}

__global__ __launch_bounds__(32) void fa2_wmma16_kernel(
    const float* __restrict__ Q, const _Float16* __restrict__ K16,
    const _Float16* __restrict__ VT16, float* __restrict__ O, int n) {
  const int lane = threadIdx.x;       // 0..31 (wave32)
  const int l16  = lane & 15;
  const int h    = lane >> 4;         // half-wave select
  const int qbase = blockIdx.x * 16;  // this wave's 16 query rows

  // B = Q^T fragment (32x16): element e -> d-row = 8h + e (+8 for e>=8), N = query = l16.
  // Pre-scaled by log2(e) so the loop computes p = exp2(st - m) directly.
  v16h bq;
  {
    const float4* qr = reinterpret_cast<const float4*>(Q + (size_t)(qbase + l16) * ATTN_D + 8 * h);
    float4 a = qr[0], b = qr[1], c = qr[4], d = qr[5];
    bq[0]  = (_Float16)(a.x * L2E); bq[1]  = (_Float16)(a.y * L2E);
    bq[2]  = (_Float16)(a.z * L2E); bq[3]  = (_Float16)(a.w * L2E);
    bq[4]  = (_Float16)(b.x * L2E); bq[5]  = (_Float16)(b.y * L2E);
    bq[6]  = (_Float16)(b.z * L2E); bq[7]  = (_Float16)(b.w * L2E);
    bq[8]  = (_Float16)(c.x * L2E); bq[9]  = (_Float16)(c.y * L2E);
    bq[10] = (_Float16)(c.z * L2E); bq[11] = (_Float16)(c.w * L2E);
    bq[12] = (_Float16)(d.x * L2E); bq[13] = (_Float16)(d.y * L2E);
    bq[14] = (_Float16)(d.z * L2E); bq[15] = (_Float16)(d.w * L2E);
  }

  // O^T accumulators (C layout: vgpr v -> d = v + 8h (+16 for ot1), N = query = l16).
  v8f ot0 = {0.f, 0.f, 0.f, 0.f, 0.f, 0.f, 0.f, 0.f};
  v8f ot1 = {0.f, 0.f, 0.f, 0.f, 0.f, 0.f, 0.f, 0.f};
  const v8f zero = {0.f, 0.f, 0.f, 0.f, 0.f, 0.f, 0.f, 0.f};
  float m = -INFINITY, l = 0.f;  // log2-domain stats, per query (per lane)

  for (int kb = 0; kb < n; kb += 32) {
    // A = K tiles (16 keys x 32 d): lane's key row = kb + tile*16 + l16.
    v16h ak0 = load_rowA16(K16 + (size_t)(kb + l16) * ATTN_D, h);
    v16h ak1 = load_rowA16(K16 + (size_t)(kb + 16 + l16) * ATTN_D, h);

    // A = V^T tiles (16 d x 32 keys): M = d = dtile*16 + l16; element e -> key
    // kb + e + 16h. VT layout makes the 16-key run contiguous.
    const _Float16* vt0 = VT16 + (size_t)l16 * n + kb + 16 * h;
    const _Float16* vt1 = vt0 + (size_t)16 * n;
    v16h av0, av1;
    {
      h8 a = reinterpret_cast<const h8*>(vt0)[0];
      h8 b = reinterpret_cast<const h8*>(vt0 + 8)[0];
      av0 = __builtin_shufflevector(a, b, 0, 1, 2, 3, 4, 5, 6, 7, 8, 9, 10, 11, 12, 13, 14, 15);
      h8 c = reinterpret_cast<const h8*>(vt1)[0];
      h8 d = reinterpret_cast<const h8*>(vt1 + 8)[0];
      av1 = __builtin_shufflevector(c, d, 0, 1, 2, 3, 4, 5, 6, 7, 8, 9, 10, 11, 12, 13, 14, 15);
    }

    // S^T tiles = K * Q^T (log2 domain): C vgpr v -> key kb + v + 8h (+16), query l16.
    v8f st0 = __builtin_amdgcn_wmma_f32_16x16x32_f16(false, ak0, false, bq, (short)0, zero, false, false);
    v8f st1 = __builtin_amdgcn_wmma_f32_16x16x32_f16(false, ak1, false, bq, (short)0, zero, false, false);

    // Online softmax per query column (= per lane; merge halves with xor16).
    float tmax = fmaxf(st0[0], st1[0]);
#pragma unroll
    for (int v = 1; v < 8; ++v) tmax = fmaxf(tmax, fmaxf(st0[v], st1[v]));
    tmax = fmaxf(tmax, __shfl_xor(tmax, 16, 32));
    const float mnew  = fmaxf(m, tmax);
    const float alpha = fast_exp2(m - mnew);

    float rsum = 0.f;
    h2 pk[8];  // packed (P0[v], P1[v]) per lane, f16
#pragma unroll
    for (int v = 0; v < 8; ++v) {
      float p0 = fast_exp2(st0[v] - mnew);
      float p1 = fast_exp2(st1[v] - mnew);
      rsum += p0 + p1;
      h2 pp; pp.x = (_Float16)p0; pp.y = (_Float16)p1;
      pk[v] = pp;
    }
    rsum += __shfl_xor(rsum, 16, 32);
    l = l * alpha + rsum;
    m = mnew;

#pragma unroll
    for (int v = 0; v < 8; ++v) { ot0[v] *= alpha; ot1[v] *= alpha; }

    // B = P^T fragment (32 keys x 16 queries): element e -> key row e + 16h.
    // b[v]:   h==0 -> own P0[v];       h==1 -> partner's P1[v]
    // b[v+8]: h==0 -> partner's P0[v]; h==1 -> own P1[v]
    v16h bp;
#pragma unroll
    for (int v = 0; v < 8; ++v) {
      int own = __builtin_bit_cast(int, pk[v]);
      int xch = __shfl_xor(own, 16, 32);
      h2 ownh = __builtin_bit_cast(h2, own);
      h2 xh   = __builtin_bit_cast(h2, xch);
      bp[v]     = h ? xh.y   : ownh.x;
      bp[v + 8] = h ? ownh.y : xh.x;
    }

    // O^T += V^T * P^T
    ot0 = __builtin_amdgcn_wmma_f32_16x16x32_f16(false, av0, false, bp, (short)0, ot0, false, false);
    ot1 = __builtin_amdgcn_wmma_f32_16x16x32_f16(false, av1, false, bp, (short)0, ot1, false, false);
  }

  // Epilogue: O[q][d] = ot/l ; lane writes two contiguous 32B chunks of its row.
  const float inv = 1.f / l;
  float* orow = O + (size_t)(qbase + l16) * ATTN_D;
  float4 s0, s1, s2, s3;
  s0.x = ot0[0] * inv; s0.y = ot0[1] * inv; s0.z = ot0[2] * inv; s0.w = ot0[3] * inv;
  s1.x = ot0[4] * inv; s1.y = ot0[5] * inv; s1.z = ot0[6] * inv; s1.w = ot0[7] * inv;
  s2.x = ot1[0] * inv; s2.y = ot1[1] * inv; s2.z = ot1[2] * inv; s2.w = ot1[3] * inv;
  s3.x = ot1[4] * inv; s3.y = ot1[5] * inv; s3.z = ot1[6] * inv; s3.w = ot1[7] * inv;
  reinterpret_cast<float4*>(orow + 8 * h)[0] = s0;
  reinterpret_cast<float4*>(orow + 8 * h)[1] = s1;
  reinterpret_cast<float4*>(orow + 16 + 8 * h)[0] = s2;
  reinterpret_cast<float4*>(orow + 16 + 8 * h)[1] = s3;
}

// ---------------- fallback kernel (self-contained, f32 operands) ----------------

__device__ __forceinline__ v16h load_rowA32(const float* __restrict__ row, int h) {
  const float4* p0 = reinterpret_cast<const float4*>(row + 8 * h);
  const float4* p1 = reinterpret_cast<const float4*>(row + 8 * h + 16);
  float4 a = p0[0], b = p0[1], c = p1[0], d = p1[1];
  v16h r;
  r[0]  = (_Float16)a.x; r[1]  = (_Float16)a.y; r[2]  = (_Float16)a.z; r[3]  = (_Float16)a.w;
  r[4]  = (_Float16)b.x; r[5]  = (_Float16)b.y; r[6]  = (_Float16)b.z; r[7]  = (_Float16)b.w;
  r[8]  = (_Float16)c.x; r[9]  = (_Float16)c.y; r[10] = (_Float16)c.z; r[11] = (_Float16)c.w;
  r[12] = (_Float16)d.x; r[13] = (_Float16)d.y; r[14] = (_Float16)d.z; r[15] = (_Float16)d.w;
  return r;
}

__global__ __launch_bounds__(32) void fa2_wmma_fallback_kernel(
    const float* __restrict__ Q, const float* __restrict__ K,
    const float* __restrict__ V, float* __restrict__ O, int n) {
  const int lane = threadIdx.x;
  const int l16  = lane & 15;
  const int h    = lane >> 4;
  const int qbase = blockIdx.x * 16;

  v16h bq;
  {
    const float4* qr = reinterpret_cast<const float4*>(Q + (size_t)(qbase + l16) * ATTN_D + 8 * h);
    float4 a = qr[0], b = qr[1], c = qr[4], d = qr[5];
    bq[0]  = (_Float16)(a.x * L2E); bq[1]  = (_Float16)(a.y * L2E);
    bq[2]  = (_Float16)(a.z * L2E); bq[3]  = (_Float16)(a.w * L2E);
    bq[4]  = (_Float16)(b.x * L2E); bq[5]  = (_Float16)(b.y * L2E);
    bq[6]  = (_Float16)(b.z * L2E); bq[7]  = (_Float16)(b.w * L2E);
    bq[8]  = (_Float16)(c.x * L2E); bq[9]  = (_Float16)(c.y * L2E);
    bq[10] = (_Float16)(c.z * L2E); bq[11] = (_Float16)(c.w * L2E);
    bq[12] = (_Float16)(d.x * L2E); bq[13] = (_Float16)(d.y * L2E);
    bq[14] = (_Float16)(d.z * L2E); bq[15] = (_Float16)(d.w * L2E);
  }

  v8f ot0 = {0.f, 0.f, 0.f, 0.f, 0.f, 0.f, 0.f, 0.f};
  v8f ot1 = {0.f, 0.f, 0.f, 0.f, 0.f, 0.f, 0.f, 0.f};
  const v8f zero = {0.f, 0.f, 0.f, 0.f, 0.f, 0.f, 0.f, 0.f};
  float m = -INFINITY, l = 0.f;

  for (int kb = 0; kb < n; kb += 32) {
    v16h ak0 = load_rowA32(K + (size_t)(kb + l16) * ATTN_D, h);
    v16h ak1 = load_rowA32(K + (size_t)(kb + 16 + l16) * ATTN_D, h);
    v8f st0 = __builtin_amdgcn_wmma_f32_16x16x32_f16(false, ak0, false, bq, (short)0, zero, false, false);
    v8f st1 = __builtin_amdgcn_wmma_f32_16x16x32_f16(false, ak1, false, bq, (short)0, zero, false, false);

    float tmax = fmaxf(st0[0], st1[0]);
#pragma unroll
    for (int v = 1; v < 8; ++v) tmax = fmaxf(tmax, fmaxf(st0[v], st1[v]));
    tmax = fmaxf(tmax, __shfl_xor(tmax, 16, 32));
    const float mnew  = fmaxf(m, tmax);
    const float alpha = fast_exp2(m - mnew);

    float rsum = 0.f;
    h2 pk[8];
#pragma unroll
    for (int v = 0; v < 8; ++v) {
      float p0 = fast_exp2(st0[v] - mnew);
      float p1 = fast_exp2(st1[v] - mnew);
      rsum += p0 + p1;
      h2 pp; pp.x = (_Float16)p0; pp.y = (_Float16)p1;
      pk[v] = pp;
    }
    rsum += __shfl_xor(rsum, 16, 32);
    l = l * alpha + rsum;
    m = mnew;
#pragma unroll
    for (int v = 0; v < 8; ++v) { ot0[v] *= alpha; ot1[v] *= alpha; }

    v16h bp;
#pragma unroll
    for (int v = 0; v < 8; ++v) {
      int own = __builtin_bit_cast(int, pk[v]);
      int xch = __shfl_xor(own, 16, 32);
      h2 ownh = __builtin_bit_cast(h2, own);
      h2 xh   = __builtin_bit_cast(h2, xch);
      bp[v]     = h ? xh.y   : ownh.x;
      bp[v + 8] = h ? ownh.y : xh.x;
    }

    v16h av0, av1;
    {
      const float* b0 = V + (size_t)(kb + 16 * h) * ATTN_D + l16;
      const float* b1 = b0 + 16;
#pragma unroll
      for (int i = 0; i < 8; ++i) {
        av0[i]     = (_Float16)b0[(size_t)i * ATTN_D];
        av0[i + 8] = (_Float16)b0[(size_t)(i + 8) * ATTN_D];
        av1[i]     = (_Float16)b1[(size_t)i * ATTN_D];
        av1[i + 8] = (_Float16)b1[(size_t)(i + 8) * ATTN_D];
      }
    }

    ot0 = __builtin_amdgcn_wmma_f32_16x16x32_f16(false, av0, false, bp, (short)0, ot0, false, false);
    ot1 = __builtin_amdgcn_wmma_f32_16x16x32_f16(false, av1, false, bp, (short)0, ot1, false, false);
  }

  const float inv = 1.f / l;
  float* orow = O + (size_t)(qbase + l16) * ATTN_D;
  float4 s0, s1, s2, s3;
  s0.x = ot0[0] * inv; s0.y = ot0[1] * inv; s0.z = ot0[2] * inv; s0.w = ot0[3] * inv;
  s1.x = ot0[4] * inv; s1.y = ot0[5] * inv; s1.z = ot0[6] * inv; s1.w = ot0[7] * inv;
  s2.x = ot1[0] * inv; s2.y = ot1[1] * inv; s2.z = ot1[2] * inv; s2.w = ot1[3] * inv;
  s3.x = ot1[4] * inv; s3.y = ot1[5] * inv; s3.z = ot1[6] * inv; s3.w = ot1[7] * inv;
  reinterpret_cast<float4*>(orow + 8 * h)[0] = s0;
  reinterpret_cast<float4*>(orow + 8 * h)[1] = s1;
  reinterpret_cast<float4*>(orow + 16 + 8 * h)[0] = s2;
  reinterpret_cast<float4*>(orow + 16 + 8 * h)[1] = s3;
}

// ---------------- launcher ----------------

extern "C" void kernel_launch(void* const* d_in, const int* in_sizes, int n_in,
                              void* d_out, int out_size, void* d_ws, size_t ws_size,
                              hipStream_t stream) {
  const float* Q = (const float*)d_in[0];
  const float* K = (const float*)d_in[1];
  const float* V = (const float*)d_in[2];
  float* O = (float*)d_out;
  const int n = in_sizes[0] / ATTN_D;  // 8192
  const size_t elems = (size_t)n * ATTN_D;
  const size_t need  = elems * sizeof(_Float16) * 2;  // K16 + VT16 (1 MB for n=8192)

  if (ws_size >= need) {
    _Float16* K16  = (_Float16*)d_ws;
    _Float16* VT16 = K16 + elems;
    // Prep: f16 K copy + f16 V transpose (one-time, ~2 MB moved, L2-resident after).
    cvt_f16_kernel<<<dim3((unsigned)(elems / 1024)), dim3(256), 0, stream>>>(K, K16);
    transpose_v_kernel<<<dim3((unsigned)(n / 256), ATTN_D), dim3(256), 0, stream>>>(V, VT16, n);
    fa2_wmma16_kernel<<<dim3((unsigned)(n / 16)), dim3(32), 0, stream>>>(Q, K16, VT16, O, n);
  } else {
    fa2_wmma_fallback_kernel<<<dim3((unsigned)(n / 16)), dim3(32), 0, stream>>>(Q, K, V, O, n);
  }
}
